// BertSelfShiftedLocalAttention_58969900974661
// MI455X (gfx1250) — matrix operand order repaired and checked by reading, not compiled, not run on hardware
//
#include <hip/hip_runtime.h>
#include <hip/hip_bf16.h>

// Problem constants (match reference)
#define B_     4
#define S_     8256
#define H_     768
#define NH_    12
#define HD_    64
#define W_     128
#define SHIFT_ 64
#define NW_    64
#define M_     (B_*S_)   // 33024 tokens

typedef __attribute__((ext_vector_type(4)))  __bf16 v4bf;
typedef __attribute__((ext_vector_type(8)))  __bf16 v8bf;
typedef __attribute__((ext_vector_type(16))) __bf16 v16bf;
typedef __attribute__((ext_vector_type(8)))  float  v8f;
typedef __attribute__((vector_size(4 * sizeof(int)))) int v4i;

#if __has_builtin(__builtin_amdgcn_global_load_async_to_lds_b128)
#define HAVE_ASYNC_LDS 1
#endif

__device__ __forceinline__ v8f vzero8() {
  v8f z;
#pragma unroll
  for (int i = 0; i < 8; ++i) z[i] = 0.0f;
  return z;
}

// 16B global->LDS copy: async DMA (ASYNCcnt) on gfx1250, sync fallback otherwise.
__device__ __forceinline__ void copy_b128_to_lds(__bf16* lds, const __bf16* g) {
#ifdef HAVE_ASYNC_LDS
  auto gp = (__attribute__((address_space(1))) v4i*)(uintptr_t)g;
  auto lp = (__attribute__((address_space(3))) v4i*)(uintptr_t)lds;
  __builtin_amdgcn_global_load_async_to_lds_b128(gp, lp, 0, 0);
#else
  *(v8bf*)lds = *(const v8bf*)g;
#endif
}

__device__ __forceinline__ void wait_async_lds() {
#ifdef HAVE_ASYNC_LDS
#if __has_builtin(__builtin_amdgcn_s_wait_asynccnt)
  __builtin_amdgcn_s_wait_asynccnt(0);
#else
  asm volatile("s_wait_asynccnt 0x0" ::: "memory");
#endif
#endif
}

// Build a 16-element bf16 fragment from two 16-byte chunks (ds_load_b128 x2).
__device__ __forceinline__ v16bf frag2(const __bf16* p0, const __bf16* p1) {
  v8bf a = *(const v8bf*)p0;
  v8bf b = *(const v8bf*)p1;
  v16bf r;
#pragma unroll
  for (int i = 0; i < 8; ++i) { r[i] = a[i]; r[i + 8] = b[i]; }
  return r;
}

__device__ __forceinline__ v8f wmma_bf16(v16bf a, v16bf b, v8f c) {
  // D = A(16x32) * B(32x16) + C, fp32 accumulate
  return __builtin_amdgcn_wmma_f32_16x16x32_bf16(false, a, false, b, (short)0, c,
                                                 false, false);
}

// ---------------------------------------------------------------- fp32->bf16
__global__ __launch_bounds__(256) void cvt_f32_bf16(const float* __restrict__ src,
                                                    __bf16* __restrict__ dst,
                                                    int n4) {
  int i = blockIdx.x * blockDim.x + threadIdx.x;
  if (i < n4) {
    const float4 v = *(const float4*)(src + (size_t)i * 4);
    v4bf o;
    o[0] = (__bf16)v.x; o[1] = (__bf16)v.y; o[2] = (__bf16)v.z; o[3] = (__bf16)v.w;
    *(v4bf*)(dst + (size_t)i * 4) = o;
  }
}

// ------------------------------------------------- fused QKV GEMM + bias + RoPE
// C[m,n] = sum_k Hb[m,k] * W[n,k]  (h @ W^T), tiles 128x128x32, 8 waves,
// double-buffered LDS with async global->LDS staging.
// Epilogue writes bf16 into QKVb[3][B][NH][S][HD] with RoPE applied to Q,K.
__global__ __launch_bounds__(256) void qkv_gemm(const __bf16* __restrict__ Hb,
                                                const __bf16* __restrict__ Wb,
                                                const float* __restrict__ bq,
                                                const float* __restrict__ bk,
                                                const float* __restrict__ bv,
                                                __bf16* __restrict__ QKVb) {
  constexpr int LDT = 48;  // padded bf16 stride (96B, 16B-aligned rows)
  __shared__ __bf16 sA[2][128 * LDT];
  __shared__ __bf16 sB[2][128 * LDT];

  const int tid  = threadIdx.x;
  const int lane = tid & 31;
  const int wave = tid >> 5;
  const int lh   = lane >> 4;   // lane half (K-split select)
  const int lr   = lane & 15;   // row/col within 16

  const int m0  = blockIdx.x * 128;
  const int n0  = blockIdx.y * 128;
  const int mat = blockIdx.z;  // 0=Q 1=K 2=V
  const __bf16* Wsel = Wb + (size_t)mat * H_ * H_;
  const float*  bias = (mat == 0) ? bq : ((mat == 1) ? bk : bv);

  v8f acc[8];
#pragma unroll
  for (int j = 0; j < 8; ++j) acc[j] = vzero8();

  // per-thread staging coordinates: 512 chunks of 8 bf16 per tile, 2 per thread
  const int c0row = tid >> 2, c0col = (tid & 3) * 8;
  const int c1row = (tid + 256) >> 2, c1col = ((tid + 256) & 3) * 8;

  auto stage = [&](int buf, int k0) {
    copy_b128_to_lds(&sA[buf][c0row * LDT + c0col],
                     &Hb[(size_t)(m0 + c0row) * H_ + k0 + c0col]);
    copy_b128_to_lds(&sA[buf][c1row * LDT + c1col],
                     &Hb[(size_t)(m0 + c1row) * H_ + k0 + c1col]);
    copy_b128_to_lds(&sB[buf][c0row * LDT + c0col],
                     &Wsel[(size_t)(n0 + c0row) * H_ + k0 + c0col]);
    copy_b128_to_lds(&sB[buf][c1row * LDT + c1col],
                     &Wsel[(size_t)(n0 + c1row) * H_ + k0 + c1col]);
  };

  stage(0, 0);
  wait_async_lds();
  __syncthreads();

  constexpr int KT = H_ / 32;  // 24
  for (int kt = 0; kt < KT; ++kt) {
    const int cur = kt & 1;
    if (kt + 1 < KT) stage(cur ^ 1, (kt + 1) * 32);  // DMA next tile during compute

    // A fragment: row = wave*16+lr, K chunks {lh*8, 16+lh*8}
    const __bf16* ar = &sA[cur][(wave * 16 + lr) * LDT];
    v16bf a = frag2(ar + lh * 8, ar + 16 + lh * 8);
#pragma unroll
    for (int j = 0; j < 8; ++j) {
      // B fragment: col row = j*16+lr, K chunks {lh*8, 16+lh*8}
      const __bf16* br = &sB[cur][(j * 16 + lr) * LDT + lh * 8];
      acc[j] = wmma_bf16(a, frag2(br, br + 16), acc[j]);
    }
    wait_async_lds();   // my async writes into the other buffer are complete
    __syncthreads();    // everyone done reading cur / writing next
  }

  // Epilogue: bias + RoPE. Pair (d, d+32) = (tile j, tile j+2), same lane/row.
#pragma unroll
  for (int jj = 0; jj < 4; ++jj) {
    const int j = (jj < 2) ? jj : jj + 2;  // 0,1,4,5
#pragma unroll
    for (int r = 0; r < 8; ++r) {
      const int m  = m0 + wave * 16 + lh * 8 + r;
      const int bI = m / S_;
      const int s  = m - bI * S_;
      const int n1 = n0 + j * 16 + lr;
      const int n2 = n1 + 32;
      float x1 = acc[j][r]     + bias[n1];
      float x2 = acc[j + 2][r] + bias[n2];
      const int he = n1 / HD_;
      const int d1 = n1 % HD_;  // in [0,32)
      const size_t base =
          (((size_t)mat * B_ + bI) * NH_ + he) * (size_t)S_ * HD_ + (size_t)s * HD_;
      if (mat < 2) {
        const int p = (s < SHIFT_) ? s : ((s - SHIFT_) & (W_ - 1));
        // inv_freq = 10000^(-2*d1/64)
        const float freq = __expf((float)d1 * (-2.0f * 9.2103403719761836f / 64.0f));
        float sn, cs;
        __sincosf((float)p * freq, &sn, &cs);
        QKVb[base + d1]      = (__bf16)(x1 * cs - x2 * sn);
        QKVb[base + d1 + 32] = (__bf16)(x2 * cs + x1 * sn);
      } else {
        QKVb[base + d1]      = (__bf16)x1;
        QKVb[base + d1 + 32] = (__bf16)x2;
      }
    }
  }
}

// ------------------------------------------------------- segment attention
// One block = one (batch, segment, head). L in {64,128}. softmax(QK^T/8) V.
template <int L, int NWAVES>
__global__ __launch_bounds__(NWAVES * 32) void attn_kernel(
    const __bf16* __restrict__ Qm, const __bf16* __restrict__ Km,
    const __bf16* __restrict__ Vm, float* __restrict__ out, int nseg, int s_off) {
  constexpr int NTH  = NWAVES * 32;
  constexpr int NT_S = L / 16;   // score col tiles
  constexpr int KS_C = L / 32;   // K-steps for P*V
  constexpr int LDK  = HD_ + 16; // 80
  constexpr int LDV  = L + 16;
  constexpr int LDP  = L + 16;

  __shared__ __bf16 smem0[L * LDP];     // sK first, then aliased as sP
  __shared__ __bf16 smemV[HD_ * LDV];   // V transposed [d][t]

  const int tid  = threadIdx.x;
  const int lane = tid & 31;
  const int wave = tid >> 5;
  const int lh   = lane >> 4;
  const int lr   = lane & 15;

  const int x   = blockIdx.x;
  const int he  = x % NH_;
  const int seg = (x / NH_) % nseg;
  const int b   = x / (NH_ * nseg);
  const int s0  = s_off + seg * L;

  const size_t rowbase =
      ((size_t)b * NH_ + he) * (size_t)S_ * HD_ + (size_t)s0 * HD_;

  __bf16* sK = smem0;
  for (int c = tid; c < L * 8; c += NTH) {       // K: async DMA, 16B chunks
    int row = c >> 3, col = (c & 7) * 8;
    copy_b128_to_lds(&sK[row * LDK + col], &Km[rowbase + (size_t)row * HD_ + col]);
  }
  for (int c = tid; c < L * 8; c += NTH) {       // V: b128 global load, 2B scatter
    int t = c >> 3, d0 = (c & 7) * 8;
    v8bf v = *(const v8bf*)&Vm[rowbase + (size_t)t * HD_ + d0];
#pragma unroll
    for (int i = 0; i < 8; ++i) smemV[(d0 + i) * LDV + t] = v[i];
  }
  wait_async_lds();
  __syncthreads();

  // Q A-fragments straight from global (each wave owns rows [wave*16, +16))
  const __bf16* qrow = Qm + rowbase + (size_t)(wave * 16 + lr) * HD_;
  v16bf aq0 = frag2(qrow + lh * 8,      qrow + 16 + lh * 8);
  v16bf aq1 = frag2(qrow + 32 + lh * 8, qrow + 48 + lh * 8);

  // scores S = Q K^T
  v8f accS[NT_S];
#pragma unroll
  for (int nt = 0; nt < NT_S; ++nt) accS[nt] = vzero8();
#pragma unroll
  for (int nt = 0; nt < NT_S; ++nt) {
    const __bf16* kr = &sK[(nt * 16 + lr) * LDK + lh * 8];
    accS[nt] = wmma_bf16(aq0, frag2(kr, kr + 16), accS[nt]);
    const __bf16* kr2 = kr + 32;
    accS[nt] = wmma_bf16(aq1, frag2(kr2, kr2 + 16), accS[nt]);
  }

  // softmax over each row: row lives in 16 lanes of one half -> xor 1,2,4,8
  const float scale = 0.125f;  // 1/sqrt(64)
  float mx[8], inv[8];
#pragma unroll
  for (int r = 0; r < 8; ++r) {
    float v = -3.0e38f;
#pragma unroll
    for (int nt = 0; nt < NT_S; ++nt) {
      accS[nt][r] *= scale;
      v = fmaxf(v, accS[nt][r]);
    }
#pragma unroll
    for (int off = 1; off < 16; off <<= 1) v = fmaxf(v, __shfl_xor(v, off, 32));
    mx[r] = v;
  }
#pragma unroll
  for (int r = 0; r < 8; ++r) {
    float sum = 0.0f;
#pragma unroll
    for (int nt = 0; nt < NT_S; ++nt) {
      float e = __expf(accS[nt][r] - mx[r]);
      accS[nt][r] = e;
      sum += e;
    }
#pragma unroll
    for (int off = 1; off < 16; off <<= 1) sum += __shfl_xor(sum, off, 32);
    inv[r] = 1.0f / sum;
  }

  __syncthreads();              // all sK reads done; reuse region for P
  __bf16* sP = smem0;
#pragma unroll
  for (int nt = 0; nt < NT_S; ++nt)
#pragma unroll
    for (int r = 0; r < 8; ++r)
      sP[(wave * 16 + lh * 8 + r) * LDP + nt * 16 + lr] =
          (__bf16)(accS[nt][r] * inv[r]);
  __syncthreads();

  // ctx = P (LxL) * V (LxHD)
  v8f accC[4];
#pragma unroll
  for (int nt = 0; nt < 4; ++nt) accC[nt] = vzero8();
#pragma unroll
  for (int kk = 0; kk < KS_C; ++kk) {
    const __bf16* prow = &sP[(wave * 16 + lr) * LDP + kk * 32];
    v16bf ap = frag2(prow + lh * 8, prow + 16 + lh * 8);
#pragma unroll
    for (int nt = 0; nt < 4; ++nt) {
      const __bf16* vr = &smemV[(nt * 16 + lr) * LDV + kk * 32 + lh * 8];
      accC[nt] = wmma_bf16(ap, frag2(vr, vr + 16), accC[nt]);
    }
  }

  // store fp32 ctx to out[b][s][he*64+d]
#pragma unroll
  for (int nt = 0; nt < 4; ++nt)
#pragma unroll
    for (int r = 0; r < 8; ++r) {
      const int m = wave * 16 + lh * 8 + r;
      const int s = s0 + m;
      const int d = nt * 16 + lr;
      out[((size_t)b * S_ + s) * H_ + he * HD_ + d] = accC[nt][r];
    }
}

// ---------------------------------------------------------------- launcher
extern "C" void kernel_launch(void* const* d_in, const int* in_sizes, int n_in,
                              void* d_out, int out_size, void* d_ws, size_t ws_size,
                              hipStream_t stream) {
  (void)in_sizes; (void)n_in; (void)out_size; (void)ws_size;
  const float* h  = (const float*)d_in[0];
  // d_in[1] attention_mask: all zeros and unused by reference softmax
  const float* Wq = (const float*)d_in[2];
  const float* bq = (const float*)d_in[3];
  const float* Wk = (const float*)d_in[4];
  const float* bk = (const float*)d_in[5];
  const float* Wv = (const float*)d_in[6];
  const float* bv = (const float*)d_in[7];
  float* out = (float*)d_out;

  char* ws = (char*)d_ws;
  const size_t nMH = (size_t)M_ * H_;      // 25,362,432 elements
  const size_t nW  = (size_t)H_ * H_;      // 589,824 elements
  __bf16* Hb   = (__bf16*)ws;                              // 50.7 MB (fits L2)
  __bf16* Wb   = (__bf16*)(ws + nMH * 2);                  // 3.5 MB
  __bf16* QKVb = (__bf16*)(ws + nMH * 2 + 3 * nW * 2);     // 152 MB

  const int t = 256;
  cvt_f32_bf16<<<(int)((nMH / 4 + t - 1) / t), t, 0, stream>>>(h, Hb, (int)(nMH / 4));
  cvt_f32_bf16<<<(int)((nW / 4 + t - 1) / t), t, 0, stream>>>(Wq, Wb,          (int)(nW / 4));
  cvt_f32_bf16<<<(int)((nW / 4 + t - 1) / t), t, 0, stream>>>(Wk, Wb + nW,     (int)(nW / 4));
  cvt_f32_bf16<<<(int)((nW / 4 + t - 1) / t), t, 0, stream>>>(Wv, Wb + 2 * nW, (int)(nW / 4));

  qkv_gemm<<<dim3(M_ / 128, H_ / 128, 3), 256, 0, stream>>>(Hb, Wb, bq, bk, bv, QKVb);

  const __bf16* Qm = QKVb;
  const __bf16* Km = QKVb + nMH;
  const __bf16* Vm = QKVb + 2 * nMH;
  // global head: 48 segments of L=64, 4 waves each
  attn_kernel<64, 4><<<B_ * NH_, 128, 0, stream>>>(Qm, Km, Vm, out, 1, 0);
  // local windows: 3072 segments of L=128, 8 waves each
  attn_kernel<128, 8><<<B_ * NW_ * NH_, 256, 0, stream>>>(Qm, Km, Vm, out, NW_, SHIFT_);
}
